// GCN_87823491269060
// MI455X (gfx1250) — matrix operand order repaired and checked by reading, not compiled, not run on hardware
//
#include <hip/hip_runtime.h>
#include <hip/hip_bf16.h>
#include <math.h>

// ---------------------------------------------------------------------------
// Problem constants (match reference)
// ---------------------------------------------------------------------------
#define NN      100000      // nodes
#define NE      400000      // edges
#define CH      128         // channels
#define NG      1024        // graphs
#define ATOM_F  9
#define ATOM_V  119
#define BOND_F  3
#define BOND_V  8

typedef __attribute__((ext_vector_type(16))) _Float16 v16h;
typedef __attribute__((ext_vector_type(4)))  _Float16 v4h;
typedef __attribute__((ext_vector_type(8)))  float    v8f;

// ---------------------------------------------------------------------------
// float atomic max via sign-aware integer atomics (works for +/- and -inf init)
// ---------------------------------------------------------------------------
__device__ __forceinline__ void atomicMaxF(float* addr, float val) {
    if (val >= 0.0f) atomicMax((int*)addr, __float_as_int(val));
    else             atomicMin((unsigned int*)addr, __float_as_uint(val));
}

// ---------------------------------------------------------------------------
// WMMA GEMM:  Out[M,128] = A[M,128] @ W[128,128] (+ bias)
//
// block = 256 threads = 8 wave32. Block computes a 32-row x 128-col slab:
// each wave owns one 16-col tile and two 16-row tiles (two accumulators,
// B fragment reused across both -> 8 v_wmma per block pass).
//
// Staging:
//   1) A tile (32x128 f32) via CDNA5 async global->LDS (ASYNCcnt, no VGPR
//      round-trip), W (128x128) synchronously with transpose + f32->f16.
//   2) s_wait_asynccnt 0 + barrier, then ONE block-wide f32->f16 convert
//      pass of the A tile (amortized once, not per wave).
//   3) k-loop feeds WMMA from contiguous f16 LDS (pure ds_load_b128).
// ---------------------------------------------------------------------------
#define LDS_W_BYTES    (CH * CH * 2)        // 32768
#define LDS_AF32_BYTES (32 * CH * 4)        // 16384
#define LDS_AF16_BYTES (32 * CH * 2)        // 8192

__global__ __launch_bounds__(256)
void gemm128_wmma(const float* __restrict__ A,
                  const float* __restrict__ W,      // row-major [K=128, N=128]
                  const float* __restrict__ bias,   // [128] or nullptr
                  float* __restrict__ Out,
                  int M)
{
    __shared__ __align__(16) char lds[LDS_W_BYTES + LDS_AF32_BYTES + LDS_AF16_BYTES];
    _Float16* sWt  = (_Float16*)lds;                                  // [n*128+k]
    float*    sA32 = (float*)   (lds + LDS_W_BYTES);                  // 32 x 128 f32
    _Float16* sA   = (_Float16*)(lds + LDS_W_BYTES + LDS_AF32_BYTES); // 32 x 128 f16

    const int tid  = threadIdx.x;
    const int lane = tid & 31;
    const int wave = tid >> 5;
    const long long rowBase = (long long)blockIdx.x * 32;

    // ---- 1) async stage A tile: 32 rows x 32 float4 = 1024 chunks, 4/thread ----
    for (int c4 = tid; c4 < 32 * 32; c4 += 256) {
        int r = c4 >> 5, qo = c4 & 31;           // row-in-tile, float4-in-row
        long long row = rowBase + r;
        if (row < M) {
            unsigned lds_addr = (unsigned)LDS_W_BYTES + (unsigned)(c4 * 16);
            const float* gsrc = A + row * CH + qo * 4;
            asm volatile("global_load_async_to_lds_b128 %0, %1, off"
                         :: "v"(lds_addr), "v"(gsrc) : "memory");
        }
    }

    // ---- stage weights (transpose, f32 -> f16) ----
    for (int i = tid; i < CH * CH; i += 256) {
        int kk = i >> 7, n = i & 127;
        sWt[n * CH + kk] = (_Float16)W[i];
    }

    asm volatile("s_wait_asynccnt 0x0" ::: "memory");  // own async loads done
    __syncthreads();                                   // -> all waves' loads done

    // ---- 2) one-shot A tile convert f32 -> f16 (float4 in, 4x f16 out) ----
    for (int c4 = tid; c4 < 32 * 32; c4 += 256) {
        float4 fv = ((const float4*)sA32)[c4];
        v4h hv;
        hv[0] = (_Float16)fv.x; hv[1] = (_Float16)fv.y;
        hv[2] = (_Float16)fv.z; hv[3] = (_Float16)fv.w;
        ((v4h*)sA)[c4] = hv;
    }
    __syncthreads();

    const int colBase = wave * 16;        // 8 waves -> 8 column tiles
    const int lidx = lane & 15;           // A: row M / B,D: col N
    const int lgrp = lane >> 4;           // lane-half selector

    v8f acc0 = {}, acc1 = {};
    #pragma unroll
    for (int k0 = 0; k0 < CH; k0 += 32) {
        // B frag: lane holds column lidx; halves = contiguous K
        //   lanes 0-15: K = k0..k0+15, lanes 16-31: K = k0+16..k0+31
        v16h b;
        #pragma unroll
        for (int j = 0; j < 16; ++j)
            b[j] = sWt[(colBase + lidx) * CH + k0 + 16 * lgrp + j];

        // A frags: lane holds row lidx (+16);
        //   lanes 0-15: K = 0..15, lanes 16-31: K = 16..31
        v16h a0, a1;
        #pragma unroll
        for (int j = 0; j < 16; ++j) {
            int kk = j;
            int ka = k0 + 16 * lgrp + kk;
            a0[j] = sA[lidx * CH + ka];
            a1[j] = sA[(16 + lidx) * CH + ka];
        }
        acc0 = __builtin_amdgcn_wmma_f32_16x16x32_f16(
                   false, a0, false, b, (short)0, acc0, false, false);
        acc1 = __builtin_amdgcn_wmma_f32_16x16x32_f16(
                   false, a1, false, b, (short)0, acc1, false, false);
    }

    // D layout: VGPR r <-> M = r + 8*lgrp ; N = lidx
    const int col = colBase + lidx;
    const float bv = bias ? bias[col] : 0.0f;
    #pragma unroll
    for (int r = 0; r < 8; ++r) {
        long long row0 = rowBase + r + 8 * lgrp;
        if (row0 < M) Out[row0 * CH + col] = acc0[r] + bv;
        long long row1 = rowBase + 16 + r + 8 * lgrp;
        if (row1 < M) Out[row1 * CH + col] = acc1[r] + bv;
    }
}

// ---------------------------------------------------------------------------
// Categorical-feature embedding sum (Atom/Bond encoders)
// ---------------------------------------------------------------------------
__global__ __launch_bounds__(256)
void encode_sum(const int* __restrict__ idx, const float* __restrict__ emb,
                float* __restrict__ out, int n, int nf, int vocab)
{
    int i = blockIdx.x * blockDim.x + threadIdx.x;
    if (i >= n * CH) return;
    int row = i >> 7, c = i & 127;
    float s = 0.0f;
    for (int f = 0; f < nf; ++f) {
        int v = idx[row * nf + f];
        s += emb[((long long)f * vocab + v) * CH + c];
    }
    out[i] = s;
}

// ---------------------------------------------------------------------------
// per-layer segment-state init: agg=0, amax=-inf, denom=0
// ---------------------------------------------------------------------------
__global__ __launch_bounds__(256)
void seg_init(float* __restrict__ agg, float* __restrict__ amax,
              float* __restrict__ denom)
{
    int i = blockIdx.x * blockDim.x + threadIdx.x;
    if (i >= NN * CH) return;
    agg[i] = 0.0f;
    if ((i & 127) == 0) {
        int n = i >> 7;
        amax[n]  = -INFINITY;
        denom[n] = 0.0f;
    }
}

// ---------------------------------------------------------------------------
// alpha[e] = <q[dst], k[src]+eE[e]> / sqrt(C) ; segment max into amax[dst]
// ---------------------------------------------------------------------------
__global__ __launch_bounds__(256)
void edge_alpha(const float* __restrict__ qn, const float* __restrict__ kn,
                const float* __restrict__ eE, const int* __restrict__ src,
                const int* __restrict__ dst, float* __restrict__ alpha,
                float* __restrict__ amax)
{
    int e = blockIdx.x * blockDim.x + threadIdx.x;
    if (e >= NE) return;
    int s = src[e], d = dst[e];
    const float4* q4 = (const float4*)(qn + (long long)d * CH);
    const float4* k4 = (const float4*)(kn + (long long)s * CH);
    const float4* e4 = (const float4*)(eE + (long long)e * CH);
    float acc = 0.0f;
    #pragma unroll 8
    for (int i = 0; i < CH / 4; ++i) {
        float4 a = q4[i], b = k4[i], c = e4[i];
        acc += a.x * (b.x + c.x) + a.y * (b.y + c.y)
             + a.z * (b.z + c.z) + a.w * (b.w + c.w);
    }
    acc *= 0.08838834764831845f;  // 1/sqrt(128)
    alpha[e] = acc;
    atomicMaxF(&amax[d], acc);
}

// ---------------------------------------------------------------------------
// ea[e] = exp(alpha - amax[dst]) ; denom[dst] += ea
// ---------------------------------------------------------------------------
__global__ __launch_bounds__(256)
void edge_exp(const float* __restrict__ alpha, const float* __restrict__ amax,
              const int* __restrict__ dst, float* __restrict__ ea,
              float* __restrict__ denom)
{
    int e = blockIdx.x * blockDim.x + threadIdx.x;
    if (e >= NE) return;
    int d = dst[e];
    float v = __expf(alpha[e] - amax[d]);
    ea[e] = v;
    atomicAdd(&denom[d], v);
}

// ---------------------------------------------------------------------------
// agg[dst, c] += ea[e] * (v[src, c] + eE[e, c])   (one thread per (e,c))
// ---------------------------------------------------------------------------
__global__ __launch_bounds__(256)
void edge_agg(const float* __restrict__ ea, const float* __restrict__ vn,
              const float* __restrict__ eE, const int* __restrict__ src,
              const int* __restrict__ dst, float* __restrict__ agg)
{
    long long i = (long long)blockIdx.x * blockDim.x + threadIdx.x;
    if (i >= (long long)NE * CH) return;
    int e = (int)(i >> 7), c = (int)(i & 127);
    int s = src[e], d = dst[e];
    float val = ea[e] * (vn[(long long)s * CH + c] + eE[i]);
    atomicAdd(&agg[(long long)d * CH + c], val);
}

// ---------------------------------------------------------------------------
// h = relu?( agg / max-safe(denom) + skip )
// ---------------------------------------------------------------------------
__global__ __launch_bounds__(256)
void finalize(const float* __restrict__ agg, const float* __restrict__ denom,
              const float* __restrict__ skip, float* __restrict__ h, int relu)
{
    int i = blockIdx.x * blockDim.x + threadIdx.x;
    if (i >= NN * CH) return;
    float d = denom[i >> 7];
    d = (d == 0.0f) ? 1.0f : d;
    float v = agg[i] / d + skip[i];
    h[i] = (relu && v < 0.0f) ? 0.0f : v;
}

// ---------------------------------------------------------------------------
// pooling
// ---------------------------------------------------------------------------
__global__ __launch_bounds__(256)
void pool_init(float* __restrict__ pooled, float* __restrict__ counts)
{
    int i = blockIdx.x * blockDim.x + threadIdx.x;
    if (i < NG * CH) pooled[i] = 0.0f;
    if (i < NG)      counts[i] = 0.0f;
}

__global__ __launch_bounds__(256)
void pool_accum(const float* __restrict__ h, const int* __restrict__ batch,
                float* __restrict__ pooled, float* __restrict__ counts)
{
    int i = blockIdx.x * blockDim.x + threadIdx.x;
    if (i >= NN * CH) return;
    int n = i >> 7, c = i & 127;
    int g = batch[n];
    atomicAdd(&pooled[(long long)g * CH + c], h[i]);
    if (c == 0) atomicAdd(&counts[g], 1.0f);
}

__global__ __launch_bounds__(256)
void final_linear(const float* __restrict__ pooled,
                  const float* __restrict__ counts,
                  const float* __restrict__ lw, const float* __restrict__ lb,
                  float* __restrict__ out)
{
    int g = blockIdx.x * blockDim.x + threadIdx.x;
    if (g >= NG) return;
    float inv = 1.0f / fmaxf(counts[g], 1.0f);
    const float4* p4 = (const float4*)(pooled + (long long)g * CH);
    const float4* w4 = (const float4*)lw;
    float s = 0.0f;
    #pragma unroll 8
    for (int i = 0; i < CH / 4; ++i) {
        float4 a = p4[i], b = w4[i];
        s += a.x * b.x + a.y * b.y + a.z * b.z + a.w * b.w;
    }
    out[g] = s * inv + lb[0];
}

// ---------------------------------------------------------------------------
// host-side launcher
// ---------------------------------------------------------------------------
extern "C" void kernel_launch(void* const* d_in, const int* in_sizes, int n_in,
                              void* d_out, int out_size, void* d_ws, size_t ws_size,
                              hipStream_t stream)
{
    (void)in_sizes; (void)n_in; (void)out_size; (void)ws_size;

    // ---- inputs (setup_inputs() dict order, params flattened depth-first) ----
    const int*   x         = (const int*)  d_in[0];   // [N, 9]
    const int*   edge_idx  = (const int*)  d_in[1];   // [2, E]
    const int*   edge_attr = (const int*)  d_in[2];   // [E, 3]
    const int*   batch     = (const int*)  d_in[3];   // [N]
    const float* atom_emb  = (const float*)d_in[4];   // [9,119,128]
    const float* bond_emb  = (const float*)d_in[5];   // [3,8,128]
    const float* lin_w     = (const float*)d_in[6];   // [128,1]
    const float* lin_b     = (const float*)d_in[7];   // [1]
    // conv layers: 8 + 9*l : Wq Wk Wv We Ws bq bk bv bs
    const int* src = edge_idx;        // edge_index[0]
    const int* dst = edge_idx + NE;   // edge_index[1]

    // ---- workspace carve-up (floats) ----
    float* w = (float*)d_ws;
    size_t off = 0;
    float* e_emb  = w + off; off += (size_t)NE * CH;
    float* edge_e = w + off; off += (size_t)NE * CH;
    float* h      = w + off; off += (size_t)NN * CH;
    float* q      = w + off; off += (size_t)NN * CH;
    float* k      = w + off; off += (size_t)NN * CH;
    float* v      = w + off; off += (size_t)NN * CH;
    float* skip   = w + off; off += (size_t)NN * CH;
    float* agg    = w + off; off += (size_t)NN * CH;
    float* alpha  = w + off; off += (size_t)NE;
    float* ea     = w + off; off += (size_t)NE;
    float* amax   = w + off; off += (size_t)NN;
    float* denom  = w + off; off += (size_t)NN;
    float* pooled = w + off; off += (size_t)NG * CH;
    float* counts = w + off; off += (size_t)NG;

    const int T = 256;
    const int gNodeC = (NN * CH + T - 1) / T;                   // per (n,c)
    const int gEdgeC = (int)(((long long)NE * CH + T - 1) / T); // per (e,c)
    const int gEdge  = (NE + T - 1) / T;
    const int gGemmN = (NN + 31) / 32;                          // 32-row slabs
    const int gGemmE = (NE + 31) / 32;

    // ---- encoders ----
    encode_sum<<<gNodeC, T, 0, stream>>>(x,         atom_emb, h,     NN, ATOM_F, ATOM_V);
    encode_sum<<<gEdgeC, T, 0, stream>>>(edge_attr, bond_emb, e_emb, NE, BOND_F, BOND_V);

    // ---- 3 TransformerConv layers ----
    for (int l = 0; l < 3; ++l) {
        const float* Wq = (const float*)d_in[8 + 9 * l + 0];
        const float* Wk = (const float*)d_in[8 + 9 * l + 1];
        const float* Wv = (const float*)d_in[8 + 9 * l + 2];
        const float* We = (const float*)d_in[8 + 9 * l + 3];
        const float* Ws = (const float*)d_in[8 + 9 * l + 4];
        const float* bq = (const float*)d_in[8 + 9 * l + 5];
        const float* bk = (const float*)d_in[8 + 9 * l + 6];
        const float* bv = (const float*)d_in[8 + 9 * l + 7];
        const float* bs = (const float*)d_in[8 + 9 * l + 8];

        gemm128_wmma<<<gGemmN, T, 0, stream>>>(h,     Wq, bq,      q,      NN);
        gemm128_wmma<<<gGemmN, T, 0, stream>>>(h,     Wk, bk,      k,      NN);
        gemm128_wmma<<<gGemmN, T, 0, stream>>>(h,     Wv, bv,      v,      NN);
        gemm128_wmma<<<gGemmE, T, 0, stream>>>(e_emb, We, nullptr, edge_e, NE);
        gemm128_wmma<<<gGemmN, T, 0, stream>>>(h,     Ws, bs,      skip,   NN);

        seg_init  <<<gNodeC, T, 0, stream>>>(agg, amax, denom);
        edge_alpha<<<gEdge,  T, 0, stream>>>(q, k, edge_e, src, dst, alpha, amax);
        edge_exp  <<<gEdge,  T, 0, stream>>>(alpha, amax, dst, ea, denom);
        edge_agg  <<<gEdgeC, T, 0, stream>>>(ea, v, edge_e, src, dst, agg);
        finalize  <<<gNodeC, T, 0, stream>>>(agg, denom, skip, h, (l < 2) ? 1 : 0);
    }

    // ---- global mean pool + linear head ----
    pool_init   <<<(NG * CH + T - 1) / T, T, 0, stream>>>(pooled, counts);
    pool_accum  <<<gNodeC, T, 0, stream>>>(h, batch, pooled, counts);
    final_linear<<<(NG + T - 1) / T, T, 0, stream>>>(pooled, counts, lin_w, lin_b,
                                                     (float*)d_out);
}
